// Attention3D_69217692942745
// MI455X (gfx1250) — matrix operand order repaired and checked by reading, not compiled, not run on hardware
//
#include <hip/hip_runtime.h>
#include <math.h>

typedef _Float16 f16;
typedef _Float16 v16h __attribute__((ext_vector_type(16)));
typedef _Float16 v8h  __attribute__((ext_vector_type(8)));
typedef float    v8f  __attribute__((ext_vector_type(8)));

#define HEADS   4
#define DHEAD   32
#define C_IN    128
#define DIM     128
#define NTOK    4096                 // D*H*W = 8*16*32
#define BATCH   2
#define M_TOTAL (BATCH * NTOK)       // 8192
#define QKV_N   384                  // 3 * DIM
#define ATT_SCALE 10.0f

// ---------------------------------------------------------------------------
// WMMA helpers (CDNA5 gfx1250, wave32): D(16x16 f32) = A(16x32 f16) * B(32x16 f16) + C
// ---------------------------------------------------------------------------
__device__ inline v8f wmma16(v16h a, v16h b, v8f c) {
    // (neg_a, A, neg_b, B, c_mod, C, reuse_a, reuse_b)
    return __builtin_amdgcn_wmma_f32_16x16x32_f16(false, a, false, b, (short)0, c, false, false);
}

// A-fragment: lane l holds row m = l%16; element e -> K = e + 16*(l/16)
// i.e. two contiguous 8-halfword chunks at +16*half and +16*half+8 from the row start.
__device__ inline v16h afrag(const f16* row_base, int half) {
    const v8h lo = *(const v8h*)(row_base + 16 * half);
    const v8h hi = *(const v8h*)(row_base + 16 * half + 8);
    v16h a;
#pragma unroll
    for (int i = 0; i < 8; ++i) { a[i] = lo[i]; a[8 + i] = hi[i]; }
    return a;
}

// B-fragment: lane l holds column n = l%16; element e -> K = e + 16*(l/16)
// i.e. one contiguous 16-halfword (32B) chunk. Caller passes the fully-resolved ptr.
__device__ inline v16h bfrag(const f16* p) {
    return *(const v16h*)p;
}

// ---------------------------------------------------------------------------
// CDNA5 async global->LDS copy (16B per lane), tracked by ASYNCcnt.
// Builtin signature (from compiler diagnostic): param 1 is int4* in the
// global address space; mirror for the LDS side.
// ---------------------------------------------------------------------------
typedef int v4i __attribute__((vector_size(16)));
typedef __attribute__((address_space(1))) v4i gv4i;
typedef __attribute__((address_space(3))) v4i lv4i;
typedef __attribute__((address_space(3))) char lchar;

__device__ inline void async_copy_b128(const void* gsrc, void* ldst) {
#if __has_builtin(__builtin_amdgcn_global_load_async_to_lds_b128)
    __builtin_amdgcn_global_load_async_to_lds_b128((gv4i*)gsrc, (lv4i*)ldst, 0, 0);
#else
    unsigned lds_addr = (unsigned)(size_t)(lchar*)ldst;
    asm volatile("global_load_async_to_lds_b128 %0, %1, off"
                 :: "v"(lds_addr), "v"(gsrc) : "memory");
#endif
}

__device__ inline void wait_async0() {
#if __has_builtin(__builtin_amdgcn_s_wait_asynccnt)
    __builtin_amdgcn_s_wait_asynccnt(0);
#else
    asm volatile("s_wait_asynccnt 0" ::: "memory");
#endif
}

// ---------------------------------------------------------------------------
// Kernel 0: convert + transpose weights to f16
//   wqt[n][k] = (f16) w_qkv[k][n]   (384 x 128)
//   wt [n][k] = (f16) w_out[k][n]   (128 x 128)
// ---------------------------------------------------------------------------
__global__ void wconv_kernel(const float* __restrict__ w_qkv,
                             const float* __restrict__ w_out,
                             f16* __restrict__ wqt, f16* __restrict__ wt) {
    int t = blockIdx.x * blockDim.x + threadIdx.x;   // 0 .. 65535
    if (t < QKV_N * C_IN) {
        int n = t >> 7, k = t & 127;
        wqt[t] = (f16)w_qkv[(size_t)k * QKV_N + n];
    } else {
        int j = t - QKV_N * C_IN;
        int n = j >> 7, k = j & 127;
        wt[j] = (f16)w_out[(size_t)k * DIM + n];
    }
}

// ---------------------------------------------------------------------------
// Kernel 1: qkv_raw[m][n] = sum_k x[m][k] * w_qkv[k][n]   (f32 out, f16 WMMA)
// grid.x = M_TOTAL/16 = 512 blocks of 128 threads (4 waves).
// ---------------------------------------------------------------------------
__global__ __launch_bounds__(128) void qkv_kernel(const float* __restrict__ x,
                                                  const f16* __restrict__ wqt,
                                                  float* __restrict__ qkv) {
    __shared__ __align__(32) f16 lx[16][C_IN];     // 4 KB: 16-row x strip in f16
    const int t  = threadIdx.x;
    const int m0 = blockIdx.x * 16;
    {   // cooperative stage + convert: each thread does 16 contiguous elements
        int row = t >> 3, c0 = (t & 7) * 16;
        const float* src = x + (size_t)(m0 + row) * C_IN + c0;
#pragma unroll
        for (int i = 0; i < 16; ++i) lx[row][c0 + i] = (f16)src[i];
    }
    __syncthreads();

    const int lane = t & 31, wv = t >> 5;
    const int half = lane >> 4, l16 = lane & 15;

    for (int nt = wv; nt < QKV_N / 16; nt += 4) {  // 24 column tiles over 4 waves
        v8f c = {};
#pragma unroll
        for (int kc = 0; kc < C_IN / 32; ++kc) {
            v16h a = afrag(&lx[l16][kc * 32], half);
            v16h b = bfrag(wqt + (size_t)(nt * 16 + l16) * C_IN + kc * 32 + 16 * half);
            c = wmma16(a, b, c);
        }
#pragma unroll
        for (int r = 0; r < 8; ++r) {
            int row = r + 8 * half;
            qkv[(size_t)(m0 + row) * QKV_N + nt * 16 + l16] = c[r];
        }
    }
}

// ---------------------------------------------------------------------------
// Kernel 2: L2-normalize q,k over head dim; pack to f16:
//   qh,kh : [B*H][N][32]   (row-major, fragment friendly)
//   vt    : [B*H][32][N]   (V transposed so B-fragments are contiguous)
// ---------------------------------------------------------------------------
__global__ void pack_kernel(const float* __restrict__ qkv,
                            f16* __restrict__ qh, f16* __restrict__ kh,
                            f16* __restrict__ vt) {
    int t = blockIdx.x * blockDim.x + threadIdx.x;   // 0 .. 32767 = B*H*N
    int n = t & (NTOK - 1);
    int h = (t >> 12) & (HEADS - 1);
    int b = t >> 14;
    const float* row = qkv + (size_t)(b * NTOK + n) * QKV_N;
    const float* q = row + h * DHEAD;
    const float* k = row + DIM + h * DHEAD;
    const float* v = row + 2 * DIM + h * DHEAD;

    float sq = 0.f, sk = 0.f;
#pragma unroll
    for (int d = 0; d < DHEAD; ++d) { sq += q[d] * q[d]; sk += k[d] * k[d]; }
    float iq = 1.0f / fmaxf(sqrtf(sq), 1e-12f);
    float ik = 1.0f / fmaxf(sqrtf(sk), 1e-12f);

    int bh = b * HEADS + h;
    f16* qo = qh + (size_t)(bh * NTOK + n) * DHEAD;
    f16* ko = kh + (size_t)(bh * NTOK + n) * DHEAD;
#pragma unroll
    for (int d = 0; d < DHEAD; ++d) {
        qo[d] = (f16)(q[d] * iq);
        ko[d] = (f16)(k[d] * ik);
    }
#pragma unroll
    for (int d = 0; d < DHEAD; ++d)
        vt[((size_t)bh * DHEAD + d) * NTOK + n] = (f16)v[d];
}

// ---------------------------------------------------------------------------
// Kernel 3: flash attention per (b,h). grid = (N/128, B*H), block = 256 (8 waves).
// K/V chunks (32 keys) are async-DMA'd into double-buffered LDS ONCE per block
// (8x L2 traffic reduction vs per-wave loads) and overlap with compute.
// Each wave owns a 16-query strip: 2 score WMMAs + online softmax + 2 acc WMMAs.
// ---------------------------------------------------------------------------
__global__ __launch_bounds__(256) void attn_kernel(const f16* __restrict__ qh,
                                                   const f16* __restrict__ kh,
                                                   const f16* __restrict__ vt,
                                                   f16* __restrict__ att) {
    __shared__ __align__(32) f16 lk[2][32][32];      // K chunk: [key][d]   (2x2KB)
    __shared__ __align__(32) f16 lv[2][32][32];      // V chunk: [d][key]   (2x2KB)
    __shared__ __align__(32) f16 lds_p[8][16][32];   // per-wave P transpose (8KB)
    const int tid  = threadIdx.x;
    const int lane = tid & 31;
    const int wv   = tid >> 5;
    const int half = lane >> 4, l16 = lane & 15;
    const int bh = blockIdx.y;
    const int b = bh >> 2, h = bh & 3;
    const int q0 = blockIdx.x * 128 + wv * 16;

    const f16* qp = qh + (size_t)bh * NTOK * DHEAD;
    const f16* kp = kh + (size_t)bh * NTOK * DHEAD;
    const f16* vp = vt + (size_t)bh * DHEAD * NTOK;

    // async-copy assignment: each thread moves one 16B packet per chunk.
    // threads 0..127 -> K rows (key-major), threads 128..255 -> V rows (d-major)
    const int crow = (tid & 127) >> 2;      // 0..31
    const int cseg = (tid & 3) * 8;         // f16 offset 0,8,16,24 within 32-elem row
    const bool isV = tid >= 128;

    // Q fragment for this wave's 16-row strip (K-dim = DHEAD = 32 exactly)
    v16h aq = afrag(qp + (size_t)(q0 + l16) * DHEAD, half);

    v8f o0 = {}, o1 = {};                 // output accum: d columns 0-15 / 16-31
    float rm[8], rl[8];                   // per-row running max / sum (replicated per half)
#pragma unroll
    for (int r = 0; r < 8; ++r) { rm[r] = -INFINITY; rl[r] = 0.f; }

    // prologue: fill buffer 0 with chunk j=0
    if (!isV) async_copy_b128(kp + (size_t)crow * DHEAD + cseg, &lk[0][crow][cseg]);
    else      async_copy_b128(vp + (size_t)crow * NTOK + cseg,  &lv[0][crow][cseg]);

    for (int j = 0, it = 0; j < NTOK; j += 32, ++it) {
        const int cur = it & 1;
        wait_async0();          // my packets for chunk `it` have landed in LDS
        __syncthreads();        // everyone's landed; prev compute buffer fully consumed

        if (j + 32 < NTOK) {    // prefetch next chunk into the other buffer
            const int nxt = cur ^ 1;
            const int jn = j + 32;
            if (!isV) async_copy_b128(kp + (size_t)(jn + crow) * DHEAD + cseg, &lk[nxt][crow][cseg]);
            else      async_copy_b128(vp + (size_t)crow * NTOK + jn + cseg,   &lv[nxt][crow][cseg]);
        }

        // scores: S = (Q Kt) * scale for keys [j, j+32)  (K fragments from LDS)
        v16h bk0 = *(const v16h*)(&lk[cur][l16][16 * half]);
        v16h bk1 = *(const v16h*)(&lk[cur][16 + l16][16 * half]);
        v8f z = {};
        v8f s0 = wmma16(aq, bk0, z);
        v8f s1 = wmma16(aq, bk1, z);

#pragma unroll
        for (int r = 0; r < 8; ++r) {
            float a = s0[r] * ATT_SCALE;
            float c = s1[r] * ATT_SCALE;
            // row max across the 16 lanes of this half (C layout: row = r + 8*half)
            float t = fmaxf(a, c);
#pragma unroll
            for (int m = 1; m < 16; m <<= 1) t = fmaxf(t, __shfl_xor(t, m, 16));
            float nm = fmaxf(rm[r], t);
            float f  = __expf(rm[r] - nm);          // rescale factor (0 on first chunk)
            float p0 = __expf(a - nm);
            float p1 = __expf(c - nm);
            o0[r] *= f; o1[r] *= f;
            float ps = p0 + p1;
#pragma unroll
            for (int m = 1; m < 16; m <<= 1) ps += __shfl_xor(ps, m, 16);
            rl[r] = rl[r] * f + ps;
            rm[r] = nm;
            // stash P tile (C layout) for transpose to A layout
            int row = r + 8 * half;
            lds_p[wv][row][l16]      = (f16)p0;
            lds_p[wv][row][16 + l16] = (f16)p1;
        }
        // P as A-fragment (same-wave LDS RAW: hardware DScnt keeps order)
        v16h ap = afrag(&lds_p[wv][l16][0], half);
        // V fragments from LDS: row = d column, contiguous over keys
        v16h bv0 = *(const v16h*)(&lv[cur][l16][16 * half]);
        v16h bv1 = *(const v16h*)(&lv[cur][16 + l16][16 * half]);
        o0 = wmma16(ap, bv0, o0);
        o1 = wmma16(ap, bv1, o1);
    }

#pragma unroll
    for (int r = 0; r < 8; ++r) {
        float inv = 1.0f / rl[r];
        int n = q0 + r + 8 * half;
        size_t base = ((size_t)(b * NTOK + n)) * DIM + h * DHEAD;
        att[base + l16]      = (f16)(o0[r] * inv);
        att[base + 16 + l16] = (f16)(o1[r] * inv);
    }
}

// ---------------------------------------------------------------------------
// Kernel 4: out[m][n] = sum_k att[m][k] * w_out[k][n] + b_out[n]  (f32 out)
// grid.x = M_TOTAL/16 = 512 blocks of 128 threads (4 waves, 8 column tiles).
// ---------------------------------------------------------------------------
__global__ __launch_bounds__(128) void proj_kernel(const f16* __restrict__ att,
                                                   const f16* __restrict__ wt,
                                                   const float* __restrict__ bias,
                                                   float* __restrict__ out) {
    const int t = threadIdx.x;
    const int lane = t & 31, wv = t >> 5;
    const int half = lane >> 4, l16 = lane & 15;
    const int m0 = blockIdx.x * 16;
    const f16* arow = att + (size_t)(m0 + l16) * DIM;

    for (int nt = wv; nt < DIM / 16; nt += 4) {
        v8f c = {};
#pragma unroll
        for (int kc = 0; kc < DIM / 32; ++kc) {
            v16h a = afrag(arow + kc * 32, half);
            v16h b = bfrag(wt + (size_t)(nt * 16 + l16) * DIM + kc * 32 + 16 * half);
            c = wmma16(a, b, c);
        }
#pragma unroll
        for (int r = 0; r < 8; ++r) {
            int row = r + 8 * half;
            int col = nt * 16 + l16;
            out[(size_t)(m0 + row) * DIM + col] = c[r] + bias[col];
        }
    }
}

// ---------------------------------------------------------------------------
// Launch
// ---------------------------------------------------------------------------
extern "C" void kernel_launch(void* const* d_in, const int* in_sizes, int n_in,
                              void* d_out, int out_size, void* d_ws, size_t ws_size,
                              hipStream_t stream) {
    (void)in_sizes; (void)n_in; (void)out_size; (void)ws_size;
    const float* x     = (const float*)d_in[0];   // [2,8,16,32,128] = [8192][128]
    const float* w_qkv = (const float*)d_in[1];   // [128][384]
    const float* w_out = (const float*)d_in[2];   // [128][128]
    const float* b_out = (const float*)d_in[3];   // [128]
    float* out = (float*)d_out;

    // workspace carve-up (all offsets 256B-aligned)
    char* ws = (char*)d_ws;
    float* qkv_raw = (float*)(ws + 0);            // 8192*384*4  = 12,582,912 B
    f16*   qh      = (f16*)(ws + 12582912);       //  2,097,152 B
    f16*   kh      = (f16*)(ws + 14680064);       //  2,097,152 B
    f16*   vt      = (f16*)(ws + 16777216);       //  2,097,152 B
    f16*   att     = (f16*)(ws + 18874368);       //  2,097,152 B
    f16*   wqt     = (f16*)(ws + 20971520);       //     98,304 B
    f16*   wt      = (f16*)(ws + 21069824);       //     32,768 B

    wconv_kernel<<<(QKV_N * C_IN + DIM * DIM) / 256, 256, 0, stream>>>(w_qkv, w_out, wqt, wt);
    qkv_kernel<<<M_TOTAL / 16, 128, 0, stream>>>(x, wqt, qkv_raw);
    pack_kernel<<<(BATCH * HEADS * NTOK) / 256, 256, 0, stream>>>(qkv_raw, qh, kh, vt);
    attn_kernel<<<dim3(NTOK / 128, BATCH * HEADS), 256, 0, stream>>>(qh, kh, vt, att);
    proj_kernel<<<M_TOTAL / 16, 128, 0, stream>>>(att, wt, b_out, out);
}
